// LineAttention_27625229648604
// MI455X (gfx1250) — matrix operand order repaired and checked by reading, not compiled, not run on hardware
//
#include <hip/hip_runtime.h>
#include <hip/hip_bf16.h>
#include <math.h>

// ---------------------------------------------------------------------------
// LineAttention on MI455X (gfx1250): bf16-resident WMMA GEMM pipeline.
//  - weights pre-packed to bf16 once (no cvt in hot loops)
//  - activations carried in bf16 between GEMMs; f32 only where VALU needs it
//  - per-block kv/k/v precompute (K_indices gather moved after projection)
//  - GEMM: 32 rows x 512 cols per block; B fragments shared across two
//    row-halves -> 1.5 b128 loads per v_wmma_f32_16x16x32_bf16
// ---------------------------------------------------------------------------

#define LQ   32768   // lines
#define BB   8192    // blocks
#define MN   8       // neighbors per line
#define LIN  768
#define BIN  768
#define HH   512
#define NHD  8
#define DH   64

typedef __bf16 bf16_t;
typedef __attribute__((ext_vector_type(2)))  __bf16 v2bf;
typedef __attribute__((ext_vector_type(8)))  __bf16 v8bf;
typedef __attribute__((ext_vector_type(16))) __bf16 v16bf;
typedef __attribute__((ext_vector_type(8)))  float  v8f;

enum { EP_BIAS = 0, EP_LN = 1, EP_LN_GELU = 2, EP_LN_GELU_RES = 3 };

// ---------------------------------------------------------------------------
// f32 -> bf16 convert, 8 elements/thread (16B stores). n8 = n/8.
// ---------------------------------------------------------------------------
__global__ void __launch_bounds__(256)
cvt_bf16_kernel(const float* __restrict__ src, bf16_t* __restrict__ dst, long n8)
{
    const long i = (long)blockIdx.x * blockDim.x + threadIdx.x;
    if (i >= n8) return;
    const float* s = src + i * 8;
    v8bf o;
#pragma unroll
    for (int e = 0; e < 8; ++e) o[e] = (bf16_t)s[e];
    *(v8bf*)(dst + i * 8) = o;
}

// ---------------------------------------------------------------------------
// GEMM: out[r, col_off+n] = epilogue( A[r,:K] @ W[n,:K] + bias[n] )
// A, W bf16 row-major; f32 accumulate via v_wmma_f32_16x16x32_bf16.
// Block = 256 thr = 8 waves, computes 32 rows x 512 cols (full rows => fused
// LayerNorm via LDS). Wave w: cols [64w,64w+64) = 4 WMMA col-tiles x 2 row
// tiles; the B fragment is reused by both row tiles, A by all 4 col tiles.
// out (f32) and out16 (bf16) are each optional (nullptr to skip).
// ---------------------------------------------------------------------------
template <int EPI>
__global__ void __launch_bounds__(256)
gemm512_kernel(const bf16_t* __restrict__ A, int lda, int K,
               const bf16_t* __restrict__ W,      // [512][K] bf16 row-major
               const float* __restrict__ bias,    // [512] or nullptr
               const float* __restrict__ gamma,
               const float* __restrict__ beta,
               const float* __restrict__ resid, int ldr,
               float*  __restrict__ out,
               bf16_t* __restrict__ out16,
               int ldo, int col_off)
{
    __shared__ float lds[32 * 512];
    __shared__ float red_s[32][8];
    __shared__ float red_q[32][8];
    __shared__ float mu_s[32];
    __shared__ float rs_s[32];

    const int tid  = threadIdx.x;
    const int wave = tid >> 5;
    const int lane = tid & 31;
    const int m    = lane & 15;      // row in tile (A) / col in tile (B,D)
    const int hi   = lane >> 4;      // K-half selector (ISA 7.12.2 layouts)
    const int r0   = blockIdx.x * 32;
    const int colBase = wave * 64;

    v8f acc[2][4];
#pragma unroll
    for (int half = 0; half < 2; ++half)
#pragma unroll
        for (int j2 = 0; j2 < 4; ++j2)
#pragma unroll
            for (int e = 0; e < 8; ++e) acc[half][j2][e] = 0.0f;

    const bf16_t* arow0 = A + (size_t)(r0 + m) * lda;        // rows r0+0..15
    const bf16_t* arow1 = A + (size_t)(r0 + 16 + m) * lda;   // rows r0+16..31
    const bf16_t* wrow0 = W + (size_t)(colBase + m) * K + 16 * hi;

    for (int kk = 0; kk < K; kk += 32) {
        // A fragments 16x32: lane(m,hi) holds K=[kk+16hi,+16) contiguous
        const v8bf a0lo = *(const v8bf*)(arow0 + kk + 16 * hi);
        const v8bf a0hi = *(const v8bf*)(arow0 + kk + 16 * hi + 8);
        const v8bf a1lo = *(const v8bf*)(arow1 + kk + 16 * hi);
        const v8bf a1hi = *(const v8bf*)(arow1 + kk + 16 * hi + 8);
        const v16bf a0 = __builtin_shufflevector(a0lo, a0hi,
            0, 1, 2, 3, 4, 5, 6, 7, 8, 9, 10, 11, 12, 13, 14, 15);
        const v16bf a1 = __builtin_shufflevector(a1lo, a1hi,
            0, 1, 2, 3, 4, 5, 6, 7, 8, 9, 10, 11, 12, 13, 14, 15);
#pragma unroll
        for (int j2 = 0; j2 < 4; ++j2) {
            // B fragment 32x16: lane holds col n, 16 contiguous K at kk+16*hi
            const v16bf b = *(const v16bf*)(wrow0 + (size_t)j2 * 16 * K + kk);
            acc[0][j2] = __builtin_amdgcn_wmma_f32_16x16x32_bf16(
                false, a0, false, b, (short)0, acc[0][j2], false, false);
            acc[1][j2] = __builtin_amdgcn_wmma_f32_16x16x32_bf16(
                false, a1, false, b, (short)0, acc[1][j2], false, false);
        }
    }

    // D layout: lane holds rows (16*half + j + 8*hi), col = colBase+16*j2+m
#pragma unroll
    for (int half = 0; half < 2; ++half) {
#pragma unroll
        for (int j2 = 0; j2 < 4; ++j2) {
            const int ncol = colBase + j2 * 16 + m;
            const float bv = bias ? bias[ncol] : 0.0f;
#pragma unroll
            for (int j = 0; j < 8; ++j) {
                const int row = 16 * half + j + 8 * hi;
                const float v = acc[half][j2][j] + bv;
                if (EPI == EP_BIAS) {
                    const size_t o = (size_t)(r0 + row) * ldo + col_off + ncol;
                    if (out)   out[o]   = v;
                    if (out16) out16[o] = (bf16_t)v;
                } else {
                    lds[row * 512 + ncol] = v;
                }
            }
        }
    }

    if (EPI != EP_BIAS) {
        __syncthreads();
        {   // 8 threads per row each reduce 64 elements
            const int row = tid >> 3, sub = tid & 7;
            const float* rp = lds + row * 512 + sub * 64;
            float s = 0.0f, q = 0.0f;
#pragma unroll 8
            for (int e = 0; e < 64; ++e) { float v = rp[e]; s += v; q += v * v; }
            red_s[row][sub] = s;
            red_q[row][sub] = q;
        }
        __syncthreads();
        if (tid < 32) {
            float s = 0.0f, q = 0.0f;
#pragma unroll
            for (int e = 0; e < 8; ++e) { s += red_s[tid][e]; q += red_q[tid][e]; }
            const float mu  = s * (1.0f / 512.0f);
            const float var = q * (1.0f / 512.0f) - mu * mu;
            mu_s[tid] = mu;
            rs_s[tid] = rsqrtf(var + 1e-5f);
        }
        __syncthreads();
#pragma unroll 4
        for (int e = 0; e < 64; ++e) {
            const int idx = tid * 64 + e;          // 256*64 == 32*512
            const int row = idx >> 9, col = idx & 511;
            float v = (lds[idx] - mu_s[row]) * rs_s[row] * gamma[col] + beta[col];
            if (EPI >= EP_LN_GELU)
                v = 0.5f * v * (1.0f + erff(v * 0.70710678118654752f)); // exact GELU
            if (EPI == EP_LN_GELU_RES)
                v += resid[(size_t)(r0 + row) * ldr + col];
            const size_t o = (size_t)(r0 + row) * ldo + col_off + col;
            if (out)   out[o]   = v;
            if (out16) out16[o] = (bf16_t)v;
        }
    }
}

// ---------------------------------------------------------------------------
// Attention: one wave per line. Shuffle-reduced dot products per head,
// masked softmax in registers, ctx written directly as bf16 (GEMM consumer).
// ---------------------------------------------------------------------------
__global__ void __launch_bounds__(256)
attn_kernel(const float* __restrict__ qh,
            const float* __restrict__ kblk,
            const float* __restrict__ vblk,
            const int* __restrict__ Kidx,
            const int* __restrict__ counts,
            bf16_t* __restrict__ ctx16)
{
    const int wave = (blockIdx.x * blockDim.x + threadIdx.x) >> 5;
    const int lane = threadIdx.x & 31;
    if (wave >= LQ) return;
    const int l = wave;

    int idx[MN];
#pragma unroll
    for (int m = 0; m < MN; ++m) idx[m] = Kidx[l * MN + m];
    const int cnt = counts[l];

#pragma unroll 2
    for (int h = 0; h < NHD; ++h) {
        const int base = h * DH + lane * 2;     // 2 dims per lane (DH=64)
        const float q0 = qh[(size_t)l * HH + base];
        const float q1 = qh[(size_t)l * HH + base + 1];

        float s[MN];
#pragma unroll
        for (int m = 0; m < MN; ++m) {
            const float* kr = kblk + (size_t)idx[m] * HH + base;
            float p = q0 * kr[0] + q1 * kr[1];
#pragma unroll
            for (int off = 16; off > 0; off >>= 1)
                p += __shfl_xor(p, off, 32);
            s[m] = (m < cnt) ? p * 0.125f : -1e30f;   // /sqrt(64), mask pads
        }
        float mx = s[0];
#pragma unroll
        for (int m = 1; m < MN; ++m) mx = fmaxf(mx, s[m]);
        float w[MN], sum = 0.0f;
#pragma unroll
        for (int m = 0; m < MN; ++m) { w[m] = __expf(s[m] - mx); sum += w[m]; }
        const float inv = 1.0f / sum;

        float c0 = 0.0f, c1 = 0.0f;
#pragma unroll
        for (int m = 0; m < MN; ++m) {
            const float* vr = vblk + (size_t)idx[m] * HH + base;
            c0 += w[m] * vr[0];
            c1 += w[m] * vr[1];
        }
        v2bf c;
        c[0] = (bf16_t)(c0 * inv);
        c[1] = (bf16_t)(c1 * inv);
        *(v2bf*)(ctx16 + (size_t)l * HH + base) = c;
    }
}

// ---------------------------------------------------------------------------
// Head: sigmoid(x @ head_w + head_b), one wave per line.
// ---------------------------------------------------------------------------
__global__ void __launch_bounds__(256)
head_kernel(const float* __restrict__ x,
            const float* __restrict__ hw,
            const float* __restrict__ hb,
            float* __restrict__ out)
{
    const int wave = (blockIdx.x * blockDim.x + threadIdx.x) >> 5;
    const int lane = threadIdx.x & 31;
    if (wave >= LQ) return;
    float p = 0.0f;
#pragma unroll
    for (int j = 0; j < HH / 32; ++j) {
        const int d = lane + 32 * j;
        p += x[(size_t)wave * HH + d] * hw[d];
    }
#pragma unroll
    for (int off = 16; off > 0; off >>= 1) p += __shfl_xor(p, off, 32);
    if (lane == 0) {
        const float z = p + hb[0];
        out[wave] = 1.0f / (1.0f + __expf(-z));
    }
}

// ---------------------------------------------------------------------------
static inline void cvt_launch(const float* src, bf16_t* dst, long n, hipStream_t s)
{
    const long n8 = n / 8;
    const int  blocks = (int)((n8 + 255) / 256);
    cvt_bf16_kernel<<<dim3(blocks), dim3(256), 0, s>>>(src, dst, n8);
}

extern "C" void kernel_launch(void* const* d_in, const int* in_sizes, int n_in,
                              void* d_out, int out_size, void* d_ws, size_t ws_size,
                              hipStream_t stream) {
    const float* global_lines = (const float*)d_in[0];
    const float* block_reprs  = (const float*)d_in[1];
    const float* Wq     = (const float*)d_in[2];
    const float* Wkv    = (const float*)d_in[3];
    const float* qn_g   = (const float*)d_in[4];
    const float* qn_b   = (const float*)d_in[5];
    const float* kvn_g  = (const float*)d_in[6];
    const float* kvn_b  = (const float*)d_in[7];
    const float* in_w   = (const float*)d_in[8];   // [1536,512]
    const float* in_b   = (const float*)d_in[9];   // [1536]
    const float* out_w  = (const float*)d_in[10];
    const float* out_b  = (const float*)d_in[11];
    const float* p1_w   = (const float*)d_in[12];  // [512,1024]
    const float* p1_b   = (const float*)d_in[13];
    const float* ln1_g  = (const float*)d_in[14];
    const float* ln1_b  = (const float*)d_in[15];
    const float* p2_w   = (const float*)d_in[16];
    const float* p2_b   = (const float*)d_in[17];
    const float* ln2_g  = (const float*)d_in[18];
    const float* ln2_b  = (const float*)d_in[19];
    const float* head_w = (const float*)d_in[20];
    const float* head_b = (const float*)d_in[21];
    const int*   Kidx   = (const int*)d_in[22];
    const int*   counts = (const int*)d_in[23];
    float* out = (float*)d_out;

    // ---- workspace layout: f32 region, then bf16 region -------------------
    float* f    = (float*)d_ws;
    float* kb   = f;                                // [8192, 512]  f32 (attn)
    float* vb   = kb + (size_t)BB * HH;             // [8192, 512]  f32 (attn)
    float* qh   = vb + (size_t)BB * HH;             // [32768,512]  f32 (attn)
    float* xb   = qh + (size_t)LQ * HH;             // [32768,512]  f32 (resid)
    float* x2   = qh;                               // reuse qh (dead after attn)

    bf16_t* h        = (bf16_t*)(xb + (size_t)LQ * HH);
    bf16_t* lines16  = h;                                   // [32768, 768]
    bf16_t* blocks16 = lines16  + (size_t)LQ * LIN;         // [8192, 768]
    bf16_t* Wq16     = blocks16 + (size_t)BB * BIN;         // [512, 768]
    bf16_t* Wkv16    = Wq16     + (size_t)HH * LIN;         // [512, 768]
    bf16_t* inw16    = Wkv16    + (size_t)HH * BIN;         // [1536, 512]
    bf16_t* outw16   = inw16    + (size_t)3 * HH * HH;      // [512, 512]
    bf16_t* p1w16    = outw16   + (size_t)HH * HH;          // [512, 1024]
    bf16_t* p2w16    = p1w16    + (size_t)HH * 2 * HH;      // [512, 512]
    bf16_t* kvb16    = p2w16    + (size_t)HH * HH;          // [8192, 512]
    bf16_t* xcat16   = kvb16    + (size_t)BB * HH;          // [32768, 1024]
    bf16_t* ctx16    = xcat16   + (size_t)LQ * 2 * HH;      // [32768, 512]
    bf16_t* xb16     = ctx16    + (size_t)LQ * HH;          // [32768, 512]

    // ---- 0) one-time bf16 packing -----------------------------------------
    cvt_launch(global_lines, lines16,  (long)LQ * LIN, stream);
    cvt_launch(block_reprs,  blocks16, (long)BB * BIN, stream);
    cvt_launch(Wq,    Wq16,   (long)HH * LIN,     stream);
    cvt_launch(Wkv,   Wkv16,  (long)HH * BIN,     stream);
    cvt_launch(in_w,  inw16,  (long)3 * HH * HH,  stream);
    cvt_launch(out_w, outw16, (long)HH * HH,      stream);
    cvt_launch(p1_w,  p1w16,  (long)HH * 2 * HH,  stream);
    cvt_launch(p2_w,  p2w16,  (long)HH * HH,      stream);

    const dim3 blk(256);
    const dim3 gB(BB / 32);    // block-side GEMMs (32 rows per block)
    const dim3 gL(LQ / 32);    // line-side GEMMs

    // 1) kv_blk = LN(block_reprs @ Wkv^T)      (per-block precompute, bf16)
    gemm512_kernel<EP_LN><<<gB, blk, 0, stream>>>(
        blocks16, BIN, BIN, Wkv16, nullptr, kvn_g, kvn_b, nullptr, 0,
        nullptr, kvb16, HH, 0);
    // 2) k_blk / v_blk = kv_blk @ w{k,v}^T + b{k,v}   (f32 for attention)
    gemm512_kernel<EP_BIAS><<<gB, blk, 0, stream>>>(
        kvb16, HH, HH, inw16 + (size_t)HH * HH, in_b + HH, nullptr, nullptr,
        nullptr, 0, kb, nullptr, HH, 0);
    gemm512_kernel<EP_BIAS><<<gB, blk, 0, stream>>>(
        kvb16, HH, HH, inw16 + (size_t)2 * HH * HH, in_b + 2 * HH, nullptr, nullptr,
        nullptr, 0, vb, nullptr, HH, 0);
    // 3) q = LN(lines @ Wq^T) -> xcat16[:, 0:512]
    gemm512_kernel<EP_LN><<<gL, blk, 0, stream>>>(
        lines16, LIN, LIN, Wq16, nullptr, qn_g, qn_b, nullptr, 0,
        nullptr, xcat16, 2 * HH, 0);
    // 4) qh = q @ wq_i^T + bq   (f32 for attention)
    gemm512_kernel<EP_BIAS><<<gL, blk, 0, stream>>>(
        xcat16, 2 * HH, HH, inw16, in_b, nullptr, nullptr,
        nullptr, 0, qh, nullptr, HH, 0);
    // 5) attention -> ctx16 (bf16)
    attn_kernel<<<dim3(LQ / 8), blk, 0, stream>>>(qh, kb, vb, Kidx, counts, ctx16);
    // 6) attn_out = ctx @ out_w^T + out_b -> xcat16[:, 512:1024]
    gemm512_kernel<EP_BIAS><<<gL, blk, 0, stream>>>(
        ctx16, HH, HH, outw16, out_b, nullptr, nullptr,
        nullptr, 0, nullptr, xcat16, 2 * HH, HH);
    // 7) x = GELU(LN(xcat @ p1_w^T + p1_b))  -> xb (f32 resid) + xb16
    gemm512_kernel<EP_LN_GELU><<<gL, blk, 0, stream>>>(
        xcat16, 2 * HH, 2 * HH, p1w16, p1_b, ln1_g, ln1_b,
        nullptr, 0, xb, xb16, HH, 0);
    // 8) x2 = x + GELU(LN(x @ p2_w^T + p2_b))
    gemm512_kernel<EP_LN_GELU_RES><<<gL, blk, 0, stream>>>(
        xb16, HH, HH, p2w16, p2_b, ln2_g, ln2_b,
        xb, HH, x2, nullptr, HH, 0);
    // 9) sigmoid head
    head_kernel<<<dim3(LQ / 8), blk, 0, stream>>>(x2, head_w, head_b, out);

    (void)in_sizes; (void)n_in; (void)out_size; (void)ws_size;
}
